// GRUModel_30193620091213
// MI455X (gfx1250) — matrix operand order repaired and checked by reading, not compiled, not run on hardware
//
#include <hip/hip_runtime.h>

typedef __attribute__((ext_vector_type(16))) __bf16 v16bf;
typedef __attribute__((ext_vector_type(8)))  __bf16 v8bf;
typedef __attribute__((ext_vector_type(8)))  float  v8f;
typedef __attribute__((ext_vector_type(4)))  int    v4i;

#define B_SZ 512
#define T_SZ 256
#define I_SZ 128
#define H_SZ 512
#define G3   (3 * H_SZ)        // 1536 gate columns
#define KC   (H_SZ + I_SZ)     // 640 fused K (h | x)
#define AKC  776               // padded A-panel row: 512 h + 2*128 x + 8 pad
                               // stride = 388 dwords == 4 (mod 64) -> 2-way min conflicts
#define HFP  516               // padded hF row (floats): decorrelates lane halves
#define BM   16                // batch rows per block (WMMA M)
#define NTHR 512               // 16 waves per block
#define GSTRIDE (H_SZ * KC)    // element stride between gates in Wc (327680)
#define WS_XB_OFF (4u << 20)   // xb starts 4MB into workspace (Wc needs ~1.9MB)

#if defined(__gfx1250__) && __has_builtin(__builtin_amdgcn_global_load_async_to_lds_b128)
#define USE_ASYNC_X 1
typedef __attribute__((address_space(1))) v4i* async_gptr_t;  // global src
typedef __attribute__((address_space(3))) v4i* async_lptr_t;  // LDS dst
#else
#define USE_ASYNC_X 0
#endif

// ---------------------------------------------------------------------------
// Prep 1: pack Wc[g][k] = bf16( k<512 ? W_hh[g][k] : W_ih[g][k-512] )
// Row g of Wc is the K-run of output gate-column g (column-major B operand).
// ---------------------------------------------------------------------------
__global__ __launch_bounds__(256) void build_wc_kernel(
    const float* __restrict__ W_ih,   // [G3][I]
    const float* __restrict__ W_hh,   // [G3][H]
    __bf16* __restrict__ Wc)          // [G3][KC]
{
  int i = blockIdx.x * 256 + threadIdx.x;
  if (i >= G3 * KC) return;
  int g = i / KC;
  int k = i - g * KC;
  float v = (k < H_SZ) ? W_hh[g * H_SZ + k] : W_ih[g * I_SZ + (k - H_SZ)];
  Wc[i] = (__bf16)v;
}

// ---------------------------------------------------------------------------
// Prep 2: x fp32 -> bf16 (so the in-loop staging is a raw 16B async copy)
// ---------------------------------------------------------------------------
__global__ __launch_bounds__(256) void build_xb_kernel(
    const float* __restrict__ x, __bf16* __restrict__ xb, int n)
{
  int i = blockIdx.x * 256 + threadIdx.x;
  if (i < n) xb[i] = (__bf16)x[i];
}

__device__ __forceinline__ v8f splat8(float v) {
  v8f r;
#pragma unroll
  for (int i = 0; i < 8; ++i) r[i] = v;
  return r;
}

__device__ __forceinline__ float fast_sigmoid(float x) {
  return __frcp_rn(1.0f + __expf(-x));
}
__device__ __forceinline__ float fast_tanh(float x) {
  return 1.0f - 2.0f * __frcp_rn(__expf(2.0f * x) + 1.0f);
}

// ---------------------------------------------------------------------------
// Persistent GRU scan: one block per 16 batch rows, full T loop in-kernel.
// 16 waves; wave w owns gate columns [w*32, w*32+32) of each of the 3 gates
// -> 8 fp32 accumulators (64 VGPRs). x region of A-panel is double-buffered;
// x(t+1) is copied global->LDS asynchronously while the GEMM runs.
// ---------------------------------------------------------------------------
__global__ __launch_bounds__(NTHR) void gru_scan_kernel(
    const __bf16* __restrict__ xb,    // [B][T][I] bf16
    const __bf16* __restrict__ Wc,    // [G3][KC] bf16
    const float* __restrict__ b_ih,   // [G3]
    const float* __restrict__ b_hh,   // [G3]
    const float* __restrict__ W_lin,  // [H]
    const float* __restrict__ b_lin,  // [1]
    float* __restrict__ out)          // [B]
{
  __shared__ __bf16 Apanel[BM][AKC];  // ~24.8 KB
  __shared__ float  hF[BM][HFP];      // ~33 KB fp32 master copy of h
  __shared__ float  red[BM][32];      // final-dot reduction

  const int tid  = threadIdx.x;
  const int wave = tid >> 5;          // 0..15
  const int lane = tid & 31;
  const int l16  = lane & 15;
  const int lhi  = lane >> 4;         // lane half (WMMA operand layout)
  const int wb   = wave * 32;         // this wave's 32 columns within each gate
  const int b0   = blockIdx.x * BM;

  // h(0) = 0 in both copies
  for (int i = tid; i < BM * H_SZ; i += NTHR) {
    int r = i >> 9, c = i & (H_SZ - 1);
    hF[r][c] = 0.0f;
    Apanel[r][c] = (__bf16)0.0f;
  }
  // stage x(0) into parity-0 slot (synchronous, once)
  if (tid < 256) {
    int row = tid >> 4, ck = tid & 15;
    uint4 v = *(const uint4*)(xb + ((size_t)(b0 + row) * T_SZ) * I_SZ + ck * 8);
    *(uint4*)&Apanel[row][H_SZ + ck * 8] = v;
  }

  // Per-column biases seed the accumulators each step (D layout: a lane's
  // 8 acc floats share one output column -> splat).
  float bias_r[2], bias_z[2], bias_nh[2], bias_nx[2];
  unsigned boff[2];                   // loop-invariant per-lane element offset into Wc
#pragma unroll
  for (int tt = 0; tt < 2; ++tt) {
    int c = wb + tt * 16 + l16;       // h-column index 0..511
    bias_r[tt]  = b_ih[c] + b_hh[c];
    bias_z[tt]  = b_ih[H_SZ + c] + b_hh[H_SZ + c];
    bias_nx[tt] = b_ih[2 * H_SZ + c];             // x-side of n gate
    bias_nh[tt] = b_hh[2 * H_SZ + c];             // h-side of n gate
    boff[tt] = (unsigned)c * KC + (unsigned)(lhi * 16);
  }
  const __bf16* arow = &Apanel[l16][lhi * 16];    // A-operand per-lane base

  for (int t = 0; t < T_SZ; ++t) {
    const int p = t & 1;              // x parity slot read this step
    __syncthreads();                  // A-panel (h + x_t) fully written/visible

    // ---- kick off x(t+1) copy into the other parity slot; overlaps GEMM ----
    if (t + 1 < T_SZ && tid < 256) {
      int row = tid >> 4, ck = tid & 15;
      const __bf16* gsrc =
          xb + ((size_t)(b0 + row) * T_SZ + (t + 1)) * I_SZ + ck * 8;
      __bf16* ldst = &Apanel[row][H_SZ + (1 - p) * I_SZ + ck * 8];
#if USE_ASYNC_X
      __builtin_amdgcn_global_load_async_to_lds_b128(
          (async_gptr_t)gsrc, (async_lptr_t)ldst, 0, 0);
#else
      *(uint4*)ldst = *(const uint4*)gsrc;
#endif
    }

    v8f acc0[2], acc1[2], acc2h[2], acc2x[2];
#pragma unroll
    for (int tt = 0; tt < 2; ++tt) {
      acc0[tt]  = splat8(bias_r[tt]);
      acc1[tt]  = splat8(bias_z[tt]);
      acc2h[tt] = splat8(bias_nh[tt]);
      acc2x[tt] = splat8(bias_nx[tt]);
    }

    // ---- recurrent K (h part): ksteps 0..15, K = 512 ----
#pragma unroll 4
    for (int ks = 0; ks < 16; ++ks) {
      const int k0 = ks * 32;
      v8bf lo = *(const v8bf*)(arow + k0);
      v8bf hi = *(const v8bf*)(arow + k0 + 8);
      v16bf a;
#pragma unroll
      for (int e = 0; e < 8; ++e) { a[e] = lo[e]; a[8 + e] = hi[e]; }

      const __bf16* kbase = Wc + k0;  // uniform (SGPR) base; gate offset = imm
      v16bf b00 = *(const v16bf*)(kbase + boff[0]);
      v16bf b01 = *(const v16bf*)(kbase + boff[1]);
      v16bf b10 = *(const v16bf*)(kbase + boff[0] + 1 * GSTRIDE);
      v16bf b11 = *(const v16bf*)(kbase + boff[1] + 1 * GSTRIDE);
      v16bf b20 = *(const v16bf*)(kbase + boff[0] + 2 * GSTRIDE);
      v16bf b21 = *(const v16bf*)(kbase + boff[1] + 2 * GSTRIDE);
      acc0[0]  = __builtin_amdgcn_wmma_f32_16x16x32_bf16(false, a, false, b00, (short)0, acc0[0],  false, false);
      acc0[1]  = __builtin_amdgcn_wmma_f32_16x16x32_bf16(false, a, false, b01, (short)0, acc0[1],  false, false);
      acc1[0]  = __builtin_amdgcn_wmma_f32_16x16x32_bf16(false, a, false, b10, (short)0, acc1[0],  false, false);
      acc1[1]  = __builtin_amdgcn_wmma_f32_16x16x32_bf16(false, a, false, b11, (short)0, acc1[1],  false, false);
      acc2h[0] = __builtin_amdgcn_wmma_f32_16x16x32_bf16(false, a, false, b20, (short)0, acc2h[0], false, false);
      acc2h[1] = __builtin_amdgcn_wmma_f32_16x16x32_bf16(false, a, false, b21, (short)0, acc2h[1], false, false);
    }

    // ---- input K (x part): 4 ksteps, K = 128, from parity slot p ----
#pragma unroll
    for (int j = 0; j < 4; ++j) {
      const int xk0 = H_SZ + p * I_SZ + j * 32;   // A-panel K offset
      const int wk0 = H_SZ + j * 32;              // Wc K offset
      v8bf lo = *(const v8bf*)(arow + xk0);
      v8bf hi = *(const v8bf*)(arow + xk0 + 8);
      v16bf a;
#pragma unroll
      for (int e = 0; e < 8; ++e) { a[e] = lo[e]; a[8 + e] = hi[e]; }

      const __bf16* kbase = Wc + wk0;
      v16bf b00 = *(const v16bf*)(kbase + boff[0]);
      v16bf b01 = *(const v16bf*)(kbase + boff[1]);
      v16bf b10 = *(const v16bf*)(kbase + boff[0] + 1 * GSTRIDE);
      v16bf b11 = *(const v16bf*)(kbase + boff[1] + 1 * GSTRIDE);
      v16bf b20 = *(const v16bf*)(kbase + boff[0] + 2 * GSTRIDE);
      v16bf b21 = *(const v16bf*)(kbase + boff[1] + 2 * GSTRIDE);
      acc0[0]  = __builtin_amdgcn_wmma_f32_16x16x32_bf16(false, a, false, b00, (short)0, acc0[0],  false, false);
      acc0[1]  = __builtin_amdgcn_wmma_f32_16x16x32_bf16(false, a, false, b01, (short)0, acc0[1],  false, false);
      acc1[0]  = __builtin_amdgcn_wmma_f32_16x16x32_bf16(false, a, false, b10, (short)0, acc1[0],  false, false);
      acc1[1]  = __builtin_amdgcn_wmma_f32_16x16x32_bf16(false, a, false, b11, (short)0, acc1[1],  false, false);
      acc2x[0] = __builtin_amdgcn_wmma_f32_16x16x32_bf16(false, a, false, b20, (short)0, acc2x[0], false, false);
      acc2x[1] = __builtin_amdgcn_wmma_f32_16x16x32_bf16(false, a, false, b21, (short)0, acc2x[1], false, false);
    }

    __syncthreads();                  // every wave done reading A-panel

    // gate math + h update (each wave owns disjoint columns)
#pragma unroll
    for (int tt = 0; tt < 2; ++tt) {
      int col = wb + tt * 16 + l16;
#pragma unroll
      for (int e = 0; e < 8; ++e) {
        int m = e + lhi * 8;          // D-layout row for this lane/VGPR
        float r = fast_sigmoid(acc0[tt][e]);
        float z = fast_sigmoid(acc1[tt][e]);
        float n = fast_tanh(acc2x[tt][e] + r * acc2h[tt][e]);
        float hold = hF[m][col];
        float hnew = (1.0f - z) * n + z * hold;
        hF[m][col] = hnew;
        Apanel[m][col] = (__bf16)hnew;
      }
    }

#if USE_ASYNC_X
    if (t + 1 < T_SZ) {
#if __has_builtin(__builtin_amdgcn_s_wait_asynccnt)
      __builtin_amdgcn_s_wait_asynccnt(0);
#else
      asm volatile("s_wait_asynccnt 0" ::: "memory");
#endif
    }
#endif
  }

  __syncthreads();
  // out[b] = hT[b,:] . W_lin + b_lin  (fp32 master copy of h)
  {
    int row = tid >> 5;               // 16 rows x 32 threads
    int c32 = tid & 31;
    float p = 0.0f;
    for (int c = c32; c < H_SZ; c += 32) p += hF[row][c] * W_lin[c];
    red[row][c32] = p;
    __syncthreads();
    if (c32 == 0) {
      float s = 0.0f;
#pragma unroll
      for (int e = 0; e < 32; ++e) s += red[row][e];
      out[b0 + row] = s + b_lin[0];
    }
  }
}

// ---------------------------------------------------------------------------
extern "C" void kernel_launch(void* const* d_in, const int* in_sizes, int n_in,
                              void* d_out, int out_size, void* d_ws, size_t ws_size,
                              hipStream_t stream) {
  const float* x     = (const float*)d_in[0];
  const float* W_ih  = (const float*)d_in[1];
  const float* W_hh  = (const float*)d_in[2];
  const float* b_ih  = (const float*)d_in[3];
  const float* b_hh  = (const float*)d_in[4];
  const float* W_lin = (const float*)d_in[5];
  const float* b_lin = (const float*)d_in[6];
  float* out = (float*)d_out;

  __bf16* Wc = (__bf16*)d_ws;                          // ~1.9 MB
  __bf16* xb = (__bf16*)((char*)d_ws + WS_XB_OFF);     // ~33.5 MB

  build_wc_kernel<<<(G3 * KC + 255) / 256, 256, 0, stream>>>(W_ih, W_hh, Wc);
  const int nx = B_SZ * T_SZ * I_SZ;
  build_xb_kernel<<<(nx + 255) / 256, 256, 0, stream>>>(x, xb, nx);
  gru_scan_kernel<<<B_SZ / BM, NTHR, 0, stream>>>(xb, Wc, b_ih, b_hh, W_lin, b_lin, out);
}
